// HierarchicalDualGATv2_74929999446108
// MI455X (gfx1250) — matrix operand order repaired and checked by reading, not compiled, not run on hardware
//
#include <hip/hip_runtime.h>
#include <math.h>

// ---------------------------------------------------------------------------
// HierarchicalDualGATv2 forward for MI455X (gfx1250, wave32, WMMA).
//
//   1) frame_gat_kernel : 800 blocks, one frame-graph per workgroup, all
//      GATv2 / LN / pooling / team-GAT work staged in LDS (~64.8 KB).
//   2) gemm_wmma_kernel : generic f32-storage GEMM, f16 WMMA inner product
//      (v_wmma_f32_16x16x32_f16), 16x64 block tile, 4 waves/block.
//      M%16==0 and K%32==0 guaranteed by all call sites -> unconditional,
//      b128-vectorized fragment loads; only ragged-N stores are predicated.
//   3) lstm_dir_kernel  : persistent recurrent kernel, one block per
//      direction; each timestep does gates(16x512) = xp[t] + h(16x128) @
//      Whh^T via WMMA (M=16 == one WMMA tile; wave32 sweet spot).
//   4) ln_elu_kernel / attn_pool_kernel : small fused elementwise stages.
// ---------------------------------------------------------------------------

typedef __attribute__((ext_vector_type(16))) _Float16 v16h;
typedef __attribute__((ext_vector_type(8)))  float    v8f;

#define NP 23
#define EB 506
#define ET 529

__device__ inline float sigmoidf_(float x) { return 1.0f / (1.0f + __expf(-x)); }
__device__ inline float eluf_(float x)     { return x > 0.0f ? x : (__expf(x) - 1.0f); }

// WMMA fragment k order used here (contiguous k, not the ISA table order; 16-bit A 16x32 / B 32x16):
//   A: lane L holds row L%16, k = ((L>>4)<<4) + e
//      -> 16 consecutive K values (vectorizable as 4x float4)
//   B: lane L holds col L%16, k = ((L>>4)<<4) + e
//      -> 16 consecutive K values (vectorizable as 4x float4 when B is N x K)
__device__ inline void pack4_(v16h& v, int at, float4 f) {
    v[at + 0] = (_Float16)f.x; v[at + 1] = (_Float16)f.y;
    v[at + 2] = (_Float16)f.z; v[at + 3] = (_Float16)f.w;
}

// ---------------------------------------------------------------------------
// Generic GEMM: C(MxN) = A(MxK) @ B + bias1 + bias2, optional B transposed
// (B given as N x K row-major when transB=1).
// Requirements (satisfied by every call site): M % 16 == 0, K % 32 == 0,
// A/B 16-byte aligned with lda/ldb multiples of 4. N may be ragged.
// ---------------------------------------------------------------------------
__global__ void gemm_wmma_kernel(const float* __restrict__ A, int lda,
                                 const float* __restrict__ Bm, int ldb, int transB,
                                 float* __restrict__ C, int ldc,
                                 int M, int N, int K,
                                 const float* __restrict__ bias1,
                                 const float* __restrict__ bias2)
{
    const int lane = threadIdx.x & 31;
    const int wave = threadIdx.x >> 5;
    const int m0   = blockIdx.x * 16;
    const int n0   = blockIdx.y * 64 + wave * 16;
    const int mrow = m0 + (lane & 15);
    const int ncol = n0 + (lane & 15);
    const int ncl  = ncol < N ? ncol : N - 1;   // clamp for loads only

    const float* Arow = A + (size_t)mrow * lda;
    const int aOff  = (lane >> 4) << 4;   // 0 or 16
    const int kbOff = (lane >> 4) << 4;   // 0 or 16

    v8f acc = {};
    if (transB) {
        const float* Brow = Bm + (size_t)ncl * ldb;
        for (int k0 = 0; k0 < K; k0 += 32) {
            __builtin_prefetch(Arow + k0 + 32, 0, 1);
            __builtin_prefetch(Brow + k0 + 32, 0, 1);
            v16h a, b;
            float4 a0 = *(const float4*)(Arow + k0 + aOff);
            float4 a1 = *(const float4*)(Arow + k0 + aOff + 4);
            float4 a2 = *(const float4*)(Arow + k0 + aOff + 8);
            float4 a3 = *(const float4*)(Arow + k0 + aOff + 12);
            pack4_(a, 0, a0); pack4_(a, 4, a1); pack4_(a, 8, a2); pack4_(a, 12, a3);
            float4 b0 = *(const float4*)(Brow + k0 + kbOff);
            float4 b1 = *(const float4*)(Brow + k0 + kbOff + 4);
            float4 b2 = *(const float4*)(Brow + k0 + kbOff + 8);
            float4 b3 = *(const float4*)(Brow + k0 + kbOff + 12);
            pack4_(b, 0, b0); pack4_(b, 4, b1); pack4_(b, 8, b2); pack4_(b, 12, b3);
            acc = __builtin_amdgcn_wmma_f32_16x16x32_f16(false, a, false, b,
                                                         (short)0, acc, false, false);
        }
    } else {
        for (int k0 = 0; k0 < K; k0 += 32) {
            v16h a, b;
            float4 a0 = *(const float4*)(Arow + k0 + aOff);
            float4 a1 = *(const float4*)(Arow + k0 + aOff + 4);
            float4 a2 = *(const float4*)(Arow + k0 + aOff + 8);
            float4 a3 = *(const float4*)(Arow + k0 + aOff + 12);
            pack4_(a, 0, a0); pack4_(a, 4, a1); pack4_(a, 8, a2); pack4_(a, 12, a3);
#pragma unroll
            for (int e = 0; e < 16; ++e)
                b[e] = (_Float16)Bm[(size_t)(k0 + kbOff + e) * ldb + ncl];
            acc = __builtin_amdgcn_wmma_f32_16x16x32_f16(false, a, false, b,
                                                         (short)0, acc, false, false);
        }
    }
    const int rbase = m0 + ((lane >> 4) << 3);
#pragma unroll
    for (int v = 0; v < 8; ++v) {
        int r = rbase + v;
        if (ncol < N) {
            float val = acc[v];
            if (bias1) val += bias1[ncol];
            if (bias2) val += bias2[ncol];
            C[(size_t)r * ldc + ncol] = val;
        }
    }
}

// ---------------------------------------------------------------------------
// Row LayerNorm + ELU, in place. One block per row.
// ---------------------------------------------------------------------------
__global__ void ln_elu_kernel(float* __restrict__ X, const float* __restrict__ g,
                              const float* __restrict__ b, int D)
{
    float* x = X + (size_t)blockIdx.x * D;
    __shared__ float red[256];
    __shared__ float s_mean, s_rstd;
    float s = 0.0f;
    for (int j = threadIdx.x; j < D; j += 256) s += x[j];
    red[threadIdx.x] = s; __syncthreads();
    for (int o = 128; o > 0; o >>= 1) {
        if (threadIdx.x < o) red[threadIdx.x] += red[threadIdx.x + o];
        __syncthreads();
    }
    if (threadIdx.x == 0) s_mean = red[0] / (float)D;
    __syncthreads();
    float m = s_mean, v = 0.0f;
    for (int j = threadIdx.x; j < D; j += 256) { float d = x[j] - m; v += d * d; }
    red[threadIdx.x] = v; __syncthreads();
    for (int o = 128; o > 0; o >>= 1) {
        if (threadIdx.x < o) red[threadIdx.x] += red[threadIdx.x + o];
        __syncthreads();
    }
    if (threadIdx.x == 0) s_rstd = rsqrtf(red[0] / (float)D + 1e-5f);
    __syncthreads();
    float rs = s_rstd;
    for (int j = threadIdx.x; j < D; j += 256) {
        float y = (x[j] - m) * rs * g[j] + b[j];
        x[j] = eluf_(y);
    }
}

// ---------------------------------------------------------------------------
// Persistent BiLSTM direction kernel. blockIdx.x = direction (0 fwd, 1 bwd).
// xp already contains x@Wih^T + bih + bhh. Gates via WMMA each timestep.
// out layout (B=16, T=50, 256), this direction writes cols [dir*128, +128).
// ---------------------------------------------------------------------------
__global__ void lstm_dir_kernel(const float* __restrict__ xpF, const float* __restrict__ xpB,
                                const float* __restrict__ WhhF, const float* __restrict__ WhhB,
                                float* __restrict__ out)
{
    const int dir = blockIdx.x;
    const float* xp  = dir ? xpB  : xpF;
    const float* Whh = dir ? WhhB : WhhF;   // (512,128) row-major, used as B^T
    const int col_off = dir * 128;

    __shared__ float s_h[16 * 128];
    __shared__ float s_c[16 * 128];
    __shared__ float s_g[16 * 512];
    for (int i = threadIdx.x; i < 16 * 128; i += 256) { s_h[i] = 0.0f; s_c[i] = 0.0f; }
    __syncthreads();

    const int lane  = threadIdx.x & 31;
    const int wave  = threadIdx.x >> 5;
    const int aOff  = (lane >> 4) << 4;
    const int kbOff = (lane >> 4) << 4;
    const float* hrow = &s_h[(lane & 15) * 128];

    for (int s = 0; s < 50; ++s) {
        const int t = dir ? (49 - s) : s;
        // gates = h @ Whh^T  (M=16, K=128, N=512), 32 n-tiles over 8 waves
        for (int nt = wave; nt < 32; nt += 8) {
            const int ncol = nt * 16 + (lane & 15);
            const float* Brow = Whh + (size_t)ncol * 128;
            v8f acc = {};
#pragma unroll
            for (int k0 = 0; k0 < 128; k0 += 32) {
                v16h a, b;
                float4 a0 = *(const float4*)(hrow + k0 + aOff);
                float4 a1 = *(const float4*)(hrow + k0 + aOff + 4);
                float4 a2 = *(const float4*)(hrow + k0 + aOff + 8);
                float4 a3 = *(const float4*)(hrow + k0 + aOff + 12);
                pack4_(a, 0, a0); pack4_(a, 4, a1); pack4_(a, 8, a2); pack4_(a, 12, a3);
                float4 b0 = *(const float4*)(Brow + k0 + kbOff);
                float4 b1 = *(const float4*)(Brow + k0 + kbOff + 4);
                float4 b2 = *(const float4*)(Brow + k0 + kbOff + 8);
                float4 b3 = *(const float4*)(Brow + k0 + kbOff + 12);
                pack4_(b, 0, b0); pack4_(b, 4, b1); pack4_(b, 8, b2); pack4_(b, 12, b3);
                acc = __builtin_amdgcn_wmma_f32_16x16x32_f16(false, a, false, b,
                                                             (short)0, acc, false, false);
            }
            const int rbase = (lane >> 4) << 3;
#pragma unroll
            for (int v = 0; v < 8; ++v) {
                int r = rbase + v; // batch row
                s_g[r * 512 + ncol] = acc[v] + xp[(r * 50 + t) * 512 + ncol];
            }
        }
        __syncthreads();
        // i, f, g, o split; update c, h
        for (int i = threadIdx.x; i < 16 * 128; i += 256) {
            int bb = i >> 7, j = i & 127;
            const float* gg = &s_g[bb * 512];
            float ig = sigmoidf_(gg[j]);
            float fg = sigmoidf_(gg[128 + j]);
            float gv = tanhf(gg[256 + j]);
            float og = sigmoidf_(gg[384 + j]);
            float c = fg * s_c[i] + ig * gv;
            float h = og * tanhf(c);
            s_c[i] = c;
            s_h[i] = h;
            out[(bb * 50 + t) * 256 + col_off + j] = h;
        }
        __syncthreads();
    }
}

// ---------------------------------------------------------------------------
// Attention pooling: scores = h@W + b, softmax over T, z = sum_t w*h.
// One block per batch element.
// ---------------------------------------------------------------------------
__global__ void attn_pool_kernel(const float* __restrict__ H, const float* __restrict__ W,
                                 const float* __restrict__ bb, float* __restrict__ Z)
{
    const int b = blockIdx.x;
    __shared__ float sc[50];
    for (int t = threadIdx.x; t < 50; t += 256) {
        float acc = bb[0];
        for (int c = 0; c < 256; ++c) acc += H[(b * 50 + t) * 256 + c] * W[c];
        sc[t] = acc;
    }
    __syncthreads();
    if (threadIdx.x == 0) {
        float mx = -3.4e38f;
        for (int t = 0; t < 50; ++t) mx = fmaxf(mx, sc[t]);
        float sm = 0.0f;
        for (int t = 0; t < 50; ++t) { sc[t] = __expf(sc[t] - mx); sm += sc[t]; }
        for (int t = 0; t < 50; ++t) sc[t] /= sm;
    }
    __syncthreads();
    for (int c = threadIdx.x; c < 256; c += 256) {
        float acc = 0.0f;
        for (int t = 0; t < 50; ++t) acc += sc[t] * H[(b * 50 + t) * 256 + c];
        Z[b * 256 + c] = acc;
    }
}

// ---------------------------------------------------------------------------
// Per-frame graph stage (pconv1 -> pnorm1 -> pconv2 -> pnorm2 -> pooling ->
// team GAT -> tnorm) producing the 192-d frame feature. One block per frame.
// ---------------------------------------------------------------------------
struct GatParams {
    const float *Wl1, *bl1, *Wr1, *br1, *att1, *bias1, *We1;
    const float *g1, *b1;
    const float *Wl2, *bl2, *Wr2, *br2, *att2, *bias2, *We2;
    const float *g2, *b2;
    const float *Wlt, *blt, *Wrt, *brt, *attt, *biast;
    const float *gt, *bt;
};

__global__ void frame_gat_kernel(const float* __restrict__ X, const float* __restrict__ EA,
                                 const int* __restrict__ EI, const int* __restrict__ TEI,
                                 GatParams P, float* __restrict__ frames)
{
    __shared__ float s_xl[NP * 256];   // xl (pconv1) -> h1 (post LN)
    __shared__ float s_xr[NP * 256];   // xr (pconv1) -> agg1 -> pconv2 workspace
    __shared__ float s_ea[ET * 4];     // edge attrs -> team workspace
    __shared__ float s_lg[ET * 4];     // logits / alpha
    __shared__ float s_nh[192];        // per-(node,head) stats

    const int f   = blockIdx.x;
    const int tid = threadIdx.x;
    const float* x  = X  + f * NP * 10;
    const float* ea = EA + f * EB * 4;
    const int* src = EI;
    const int* dst = EI + ET;

    // ---- edge attrs + per-node mean (self-loop attr) ----
    for (int i = tid; i < EB * 4; i += 256) s_ea[i] = ea[i];
    __syncthreads();
    for (int i = tid; i < NP * 4; i += 256) {
        int n = i >> 2, c = i & 3;
        float sum = 0.0f, deg = 0.0f;
        for (int e = 0; e < EB; ++e)
            if (dst[e] == n) { sum += s_ea[e * 4 + c]; deg += 1.0f; }
        s_ea[(EB + n) * 4 + c] = sum / fmaxf(deg, 1.0f);
    }
    __syncthreads();

    // ---- pconv1: xl / xr (23 x 256) ----
    for (int i = tid; i < NP * 256; i += 256) {
        int n = i >> 8, j = i & 255;
        float al = P.bl1[j], ar = P.br1[j];
        for (int k = 0; k < 10; ++k) {
            float xv = x[n * 10 + k];
            al += xv * P.Wl1[k * 256 + j];
            ar += xv * P.Wr1[k * 256 + j];
        }
        s_xl[i] = al; s_xr[i] = ar;
    }
    __syncthreads();

    // ---- pconv1 logits (529 edges x 4 heads) ----
    for (int i = tid; i < ET * 4; i += 256) {
        int e = i >> 2, h = i & 3;
        int s = src[e], d = dst[e];
        float acc = 0.0f;
        for (int c = 0; c < 64; ++c) {
            int j = h * 64 + c;
            float ew = s_ea[e * 4 + 0] * P.We1[0 * 256 + j] + s_ea[e * 4 + 1] * P.We1[1 * 256 + j]
                     + s_ea[e * 4 + 2] * P.We1[2 * 256 + j] + s_ea[e * 4 + 3] * P.We1[3 * 256 + j];
            float m = s_xl[s * 256 + j] + s_xr[d * 256 + j] + ew;
            m = m > 0.0f ? m : 0.2f * m;
            acc += m * P.att1[j];
        }
        s_lg[i] = acc;
    }
    __syncthreads();
    // segment softmax over dst
    for (int i = tid; i < NP * 4; i += 256) {
        int n = i >> 2, h = i & 3;
        float mx = -3.4e38f;
        for (int e = 0; e < ET; ++e) if (dst[e] == n) mx = fmaxf(mx, s_lg[e * 4 + h]);
        float dn = 0.0f;
        for (int e = 0; e < ET; ++e) if (dst[e] == n) dn += __expf(s_lg[e * 4 + h] - mx);
        s_nh[i] = mx; s_nh[96 + i] = dn;
    }
    __syncthreads();
    for (int i = tid; i < ET * 4; i += 256) {
        int e = i >> 2, h = i & 3; int d = dst[e];
        s_lg[i] = __expf(s_lg[i] - s_nh[d * 4 + h]) / s_nh[96 + d * 4 + h];
    }
    __syncthreads();
    // aggregate (overwrites s_xr)
    for (int i = tid; i < NP * 256; i += 256) {
        int n = i >> 8, j = i & 255; int h = j >> 6;
        float acc = P.bias1[j];
        for (int e = 0; e < ET; ++e)
            if (dst[e] == n) acc += s_lg[e * 4 + h] * s_xl[src[e] * 256 + j];
        s_xr[i] = acc;
    }
    __syncthreads();
    // pnorm1 LN + ELU -> h1 into s_xl
    for (int n = tid; n < NP; n += 256) {
        float m = 0.0f;
        for (int j = 0; j < 256; ++j) m += s_xr[n * 256 + j];
        m /= 256.0f;
        float v = 0.0f;
        for (int j = 0; j < 256; ++j) { float d0 = s_xr[n * 256 + j] - m; v += d0 * d0; }
        v /= 256.0f;
        s_nh[n] = m; s_nh[32 + n] = rsqrtf(v + 1e-5f);
    }
    __syncthreads();
    for (int i = tid; i < NP * 256; i += 256) {
        int n = i >> 8, j = i & 255;
        float y = (s_xr[i] - s_nh[n]) * s_nh[32 + n] * P.g1[j] + P.b1[j];
        s_xl[i] = eluf_(y);
    }
    __syncthreads();

    // ---- pconv2: xl2/xr2 (23 x 64 each) in s_xr ----
    for (int i = tid; i < NP * 64 * 2; i += 256) {
        int which = i / (NP * 64); int r = i - which * NP * 64;
        int n = r >> 6, j = r & 63;
        const float* W = which ? P.Wr2 : P.Wl2;
        float acc = which ? P.br2[j] : P.bl2[j];
        for (int k = 0; k < 256; ++k) acc += s_xl[n * 256 + k] * W[k * 64 + j];
        s_xr[which * NP * 64 + r] = acc;
    }
    __syncthreads();
    float* xl2 = s_xr;
    float* xr2 = s_xr + NP * 64;
    for (int e = tid; e < ET; e += 256) {
        int s = src[e], d = dst[e];
        float acc = 0.0f;
        for (int c = 0; c < 64; ++c) {
            float ew = s_ea[e * 4 + 0] * P.We2[0 * 64 + c] + s_ea[e * 4 + 1] * P.We2[1 * 64 + c]
                     + s_ea[e * 4 + 2] * P.We2[2 * 64 + c] + s_ea[e * 4 + 3] * P.We2[3 * 64 + c];
            float m = xl2[s * 64 + c] + xr2[d * 64 + c] + ew;
            m = m > 0.0f ? m : 0.2f * m;
            acc += m * P.att2[c];
        }
        s_lg[e] = acc;
    }
    __syncthreads();
    for (int n = tid; n < NP; n += 256) {
        float mx = -3.4e38f;
        for (int e = 0; e < ET; ++e) if (dst[e] == n) mx = fmaxf(mx, s_lg[e]);
        float dn = 0.0f;
        for (int e = 0; e < ET; ++e) if (dst[e] == n) dn += __expf(s_lg[e] - mx);
        s_nh[n] = mx; s_nh[32 + n] = dn;
    }
    __syncthreads();
    for (int e = tid; e < ET; e += 256) {
        int d = dst[e];
        s_lg[e] = __expf(s_lg[e] - s_nh[d]) / s_nh[32 + d];
    }
    __syncthreads();
    float* h2raw = s_xr + 2 * NP * 64;
    float* h2    = s_xr + 3 * NP * 64;
    for (int i = tid; i < NP * 64; i += 256) {
        int n = i >> 6, j = i & 63;
        float acc = P.bias2[j];
        for (int e = 0; e < ET; ++e)
            if (dst[e] == n) acc += s_lg[e] * xl2[src[e] * 64 + j];
        h2raw[i] = acc;
    }
    __syncthreads();
    for (int n = tid; n < NP; n += 256) {
        float m = 0.0f;
        for (int j = 0; j < 64; ++j) m += h2raw[n * 64 + j];
        m /= 64.0f;
        float v = 0.0f;
        for (int j = 0; j < 64; ++j) { float d0 = h2raw[n * 64 + j] - m; v += d0 * d0; }
        v /= 64.0f;
        s_nh[n] = m; s_nh[32 + n] = rsqrtf(v + 1e-5f);
    }
    __syncthreads();
    for (int i = tid; i < NP * 64; i += 256) {
        int n = i >> 6, j = i & 63;
        float y = (h2raw[i] - s_nh[n]) * s_nh[32 + n] * P.g2[j] + P.b2[j];
        h2[i] = eluf_(y);
    }
    __syncthreads();

    // ---- pooling: g1 = [mean, max] over nodes ----
    for (int c = tid; c < 64; c += 256) {
        float mn = 0.0f, mx = -3.4e38f;
        for (int n = 0; n < NP; ++n) { float v = h2[n * 64 + c]; mn += v; mx = fmaxf(mx, v); }
        frames[f * 192 + c]      = mn / 23.0f;
        frames[f * 192 + 64 + c] = mx;
    }
    // team_x (3 x 64) into s_ea[0..191]
    for (int c = tid; c < 64; c += 256) {
        float a = 0.0f, b = 0.0f;
        for (int n = 0; n < 11; ++n)  a += h2[n * 64 + c];
        for (int n = 11; n < 22; ++n) b += h2[n * 64 + c];
        s_ea[c]       = a / 11.0f;
        s_ea[64 + c]  = b / 11.0f;
        s_ea[128 + c] = h2[22 * 64 + c];
    }
    __syncthreads();
    // tconv xl_t / xr_t (3 x 128 each) at s_ea[192..]
    for (int i = tid; i < 3 * 128 * 2; i += 256) {
        int which = i / 384; int r = i - which * 384;
        int n = r >> 7, j = r & 127;
        const float* W = which ? P.Wrt : P.Wlt;
        float acc = which ? P.brt[j] : P.blt[j];
        for (int k = 0; k < 64; ++k) acc += s_ea[n * 64 + k] * W[k * 128 + j];
        s_ea[192 + which * 384 + r] = acc;
    }
    __syncthreads();
    const int* tsrc = TEI;
    const int* tdst = TEI + 9;
    for (int i = tid; i < 18; i += 256) {
        int e = i >> 1, h = i & 1; int s = tsrc[e], d = tdst[e];
        float acc = 0.0f;
        for (int c = 0; c < 64; ++c) {
            int j = h * 64 + c;
            float m = s_ea[192 + s * 128 + j] + s_ea[576 + d * 128 + j];
            m = m > 0.0f ? m : 0.2f * m;
            acc += m * P.attt[j];
        }
        s_nh[100 + i] = acc;
    }
    __syncthreads();
    for (int i = tid; i < 6; i += 256) {
        int n = i >> 1, h = i & 1;
        float mx = -3.4e38f;
        for (int e = 0; e < 9; ++e) if (tdst[e] == n) mx = fmaxf(mx, s_nh[100 + e * 2 + h]);
        float dn = 0.0f;
        for (int e = 0; e < 9; ++e) if (tdst[e] == n) dn += __expf(s_nh[100 + e * 2 + h] - mx);
        s_nh[120 + i] = mx; s_nh[128 + i] = dn;
    }
    __syncthreads();
    for (int i = tid; i < 18; i += 256) {
        int e = i >> 1, h = i & 1; int d = tdst[e];
        s_nh[100 + i] = __expf(s_nh[100 + i] - s_nh[120 + d * 2 + h]) / s_nh[128 + d * 2 + h];
    }
    __syncthreads();
    for (int i = tid; i < 3 * 64; i += 256) {
        int n = i >> 6, c = i & 63;
        float a0 = 0.0f, a1 = 0.0f;
        for (int e = 0; e < 9; ++e)
            if (tdst[e] == n) {
                a0 += s_nh[100 + e * 2 + 0] * s_ea[192 + tsrc[e] * 128 + c];
                a1 += s_nh[100 + e * 2 + 1] * s_ea[192 + tsrc[e] * 128 + 64 + c];
            }
        s_ea[960 + i] = 0.5f * (a0 + a1) + P.biast[c];
    }
    __syncthreads();
    for (int n = tid; n < 3; n += 256) {
        float m = 0.0f;
        for (int j = 0; j < 64; ++j) m += s_ea[960 + n * 64 + j];
        m /= 64.0f;
        float v = 0.0f;
        for (int j = 0; j < 64; ++j) { float d0 = s_ea[960 + n * 64 + j] - m; v += d0 * d0; }
        v /= 64.0f;
        s_nh[140 + n] = m; s_nh[144 + n] = rsqrtf(v + 1e-5f);
    }
    __syncthreads();
    for (int i = tid; i < 3 * 64; i += 256) {
        int n = i >> 6, j = i & 63;
        float y = (s_ea[960 + i] - s_nh[140 + n]) * s_nh[144 + n] * P.gt[j] + P.bt[j];
        s_ea[960 + i] = eluf_(y);
    }
    __syncthreads();
    for (int c = tid; c < 64; c += 256)
        frames[f * 192 + 128 + c] =
            (s_ea[960 + c] + s_ea[960 + 64 + c] + s_ea[960 + 128 + c]) * (1.0f / 3.0f);
}

// ---------------------------------------------------------------------------
// Host launch. Param leaves assumed flattened in recursive insertion order of
// the setup_inputs() dict: x, edge_attr, edge_index, team_edge_index, then 78
// param leaves (pconv1{Wl,bl,Wr,br,att,bias,We}, pnorm1{g,b}, pconv2{...},
// pnorm2, tconv{Wl,bl,Wr,br,att,bias}, tnorm, fproj{W,b,g,bb},
// lstm{l0f,l0b,l1f,l1b}{Wih,Whh,bih,bhh}, attn{W,b},
// heads{formation,set_piece,pass_net,movement,pass_quality}{W1,b1,g,bb,W2,b2}).
// ---------------------------------------------------------------------------
extern "C" void kernel_launch(void* const* d_in, const int* in_sizes, int n_in,
                              void* d_out, int out_size, void* d_ws, size_t ws_size,
                              hipStream_t stream)
{
    (void)in_sizes; (void)n_in; (void)out_size; (void)ws_size;

    const float* x   = (const float*)d_in[0];
    const float* ea  = (const float*)d_in[1];
    const int*   ei  = (const int*)d_in[2];
    const int*   tei = (const int*)d_in[3];
    auto pr = [&](int i) { return (const float*)d_in[4 + i]; };

    // workspace layout (floats), ~9.5 MB total
    float* ws      = (float*)d_ws;
    float* frames  = ws;                       // 800*192
    float* buf_fp  = frames + 800 * 192;       // 800*192
    float* xp0f    = buf_fp + 800 * 192;       // 800*512
    float* xp0b    = xp0f + 800 * 512;         // 800*512
    float* h0      = xp0b + 800 * 512;         // 800*256
    float* xp1f    = h0 + 800 * 256;           // 800*512
    float* xp1b    = xp1f + 800 * 512;         // 800*512
    float* h1      = xp1b + 800 * 512;         // 800*256
    float* z       = h1 + 800 * 256;           // 16*256
    float* htmp    = z + 16 * 256;             // 16*128

    GatParams P;
    P.Wl1 = pr(0);  P.bl1 = pr(1);  P.Wr1 = pr(2);  P.br1 = pr(3);
    P.att1 = pr(4); P.bias1 = pr(5); P.We1 = pr(6);
    P.g1 = pr(7);   P.b1 = pr(8);
    P.Wl2 = pr(9);  P.bl2 = pr(10); P.Wr2 = pr(11); P.br2 = pr(12);
    P.att2 = pr(13); P.bias2 = pr(14); P.We2 = pr(15);
    P.g2 = pr(16);  P.b2 = pr(17);
    P.Wlt = pr(18); P.blt = pr(19); P.Wrt = pr(20); P.brt = pr(21);
    P.attt = pr(22); P.biast = pr(23);
    P.gt = pr(24);  P.bt = pr(25);

    // 1) per-frame graph stage
    frame_gat_kernel<<<800, 256, 0, stream>>>(x, ea, ei, tei, P, frames);

    // 2) fproj GEMM (800x192x192) + LN/ELU
    gemm_wmma_kernel<<<dim3(50, 3), 128, 0, stream>>>(frames, 192, pr(26), 192, 0,
                                                      buf_fp, 192, 800, 192, 192,
                                                      pr(27), nullptr);
    ln_elu_kernel<<<800, 256, 0, stream>>>(buf_fp, pr(28), pr(29), 192);

    // 3) BiLSTM layer 0: input projections (Wih^T) then persistent recurrence
    gemm_wmma_kernel<<<dim3(50, 8), 128, 0, stream>>>(buf_fp, 192, pr(30), 192, 1,
                                                      xp0f, 512, 800, 512, 192,
                                                      pr(32), pr(33));
    gemm_wmma_kernel<<<dim3(50, 8), 128, 0, stream>>>(buf_fp, 192, pr(34), 192, 1,
                                                      xp0b, 512, 800, 512, 192,
                                                      pr(36), pr(37));
    lstm_dir_kernel<<<2, 256, 0, stream>>>(xp0f, xp0b, pr(31), pr(35), h0);

    // 4) BiLSTM layer 1
    gemm_wmma_kernel<<<dim3(50, 8), 128, 0, stream>>>(h0, 256, pr(38), 256, 1,
                                                      xp1f, 512, 800, 512, 256,
                                                      pr(40), pr(41));
    gemm_wmma_kernel<<<dim3(50, 8), 128, 0, stream>>>(h0, 256, pr(42), 256, 1,
                                                      xp1b, 512, 800, 512, 256,
                                                      pr(44), pr(45));
    lstm_dir_kernel<<<2, 256, 0, stream>>>(xp1f, xp1b, pr(39), pr(43), h1);

    // 5) attention pooling
    attn_pool_kernel<<<16, 256, 0, stream>>>(h1, pr(46), pr(47), z);

    // 6) MLP heads -> d_out (16 x 77), columns per head
    struct HeadDesc { int base, hid, outc, coloff; };
    const HeadDesc H[5] = { {48, 128, 5, 0}, {54, 128, 4, 5}, {60, 128, 22, 9},
                            {66, 128, 44, 31}, {72, 64, 2, 75} };
    float* outp = (float*)d_out;
    for (int i = 0; i < 5; ++i) {
        gemm_wmma_kernel<<<dim3(1, (H[i].hid + 63) / 64), 128, 0, stream>>>(
            z, 256, pr(H[i].base), H[i].hid, 0, htmp, H[i].hid,
            16, H[i].hid, 256, pr(H[i].base + 1), nullptr);
        ln_elu_kernel<<<16, 256, 0, stream>>>(htmp, pr(H[i].base + 2), pr(H[i].base + 3),
                                              H[i].hid);
        gemm_wmma_kernel<<<dim3(1, (H[i].outc + 63) / 64), 128, 0, stream>>>(
            htmp, H[i].hid, pr(H[i].base + 4), H[i].outc, 0, outp + H[i].coloff, 77,
            16, H[i].outc, H[i].hid, pr(H[i].base + 5), nullptr);
    }
}
